// GCN_64106681860346
// MI455X (gfx1250) — matrix operand order repaired and checked by reading, not compiled, not run on hardware
//
#include <hip/hip_runtime.h>
#include <hip/hip_bf16.h>

// ---------------------------------------------------------------------------
// CDNA5 (gfx1250) GCN forward. All large GEMMs use v_wmma_f32_16x16x32_bf16.
// Activations mirrored in bf16 (xb) for the matrix pipes; accumulation f32.
// 32-row M-tiles: each packed-B fragment load feeds two WMMAs (2 M-subtiles),
// halving L2 weight traffic vs 16-row tiles.
// ---------------------------------------------------------------------------

typedef __attribute__((ext_vector_type(16))) __bf16 v16bf;
typedef __attribute__((ext_vector_type(8)))  __bf16 v8bf;
typedef __attribute__((ext_vector_type(8)))  float  v8f;

__device__ __forceinline__ __bf16 f2bf(float f) { return (__bf16)f; }

union AFrag { v16bf v; v8bf h[2]; };

// A-matrix 16x32 bf16 fragment from LDS row-major [16][ldk] at k-tile kt.
// Contiguous K order (matches packed B): lanes 0-15 (row=lane)  hold K = 0..15
//                 lanes 16-31 (row=lane-16) hold K = 16..31
__device__ __forceinline__ v16bf load_a_frag(const __bf16* As, int ldk, int kt, int lane) {
  const int row = lane & 15;
  const int kb  = (lane & 16) ? 16 : 0;
  const __bf16* p = As + row * ldk + kt * 32 + kb;
  AFrag f;
  f.h[0] = *(const v8bf*)(p);        // K = kb .. kb+7
  f.h[1] = *(const v8bf*)(p + 8);    // K = kb+8 .. kb+15
  return f.v;
}

// ---------------------------------------------------------------------------
// Weight packing: W (f32, row-major K x N) -> bf16 B-fragment-major layout
// [nt][kt][lane][16].  Lane l<16: n=nt*16+l, K = kt*32 + 0..15
//                      Lane l>=16: n=nt*16+l-16, K = kt*32 + 16..31
// ---------------------------------------------------------------------------
__global__ void pack_b_kernel(const float* __restrict__ W, __bf16* __restrict__ Wp,
                              int K, int Nn) {
  int idx = blockIdx.x * blockDim.x + threadIdx.x;
  int total = (Nn / 16) * (K / 32) * 32;
  if (idx >= total) return;
  int lane = idx & 31;
  int kt   = (idx >> 5) % (K / 32);
  int nt   = (idx >> 5) / (K / 32);
  int n    = nt * 16 + (lane & 15);
  int kg   = (lane & 16) ? 16 : 0;
  int kbase = kt * 32 + kg;
  __bf16* outp = Wp + (size_t)idx * 16;
#pragma unroll
  for (int i = 0; i < 16; ++i) outp[i] = f2bf(W[(size_t)(kbase + i) * Nn + n]);
}

// ---------------------------------------------------------------------------
// Prep: feat = relu(inputs@W1+b1)@W2+b2 + pe[pos];  write x f32 and xb bf16.
// ---------------------------------------------------------------------------
__global__ __launch_bounds__(64) void prep_kernel(
    const float* __restrict__ inputs, const float* __restrict__ pe,
    const float* __restrict__ W1, const float* __restrict__ b1,
    const float* __restrict__ W2, const float* __restrict__ b2,
    float* __restrict__ x, __bf16* __restrict__ xb, int n_nodes) {
  __shared__ float fin[64];
  __shared__ float hid[64];
  int node = blockIdx.x;
  if (node >= n_nodes) return;
  int t = threadIdx.x;
  fin[t] = inputs[(size_t)node * 64 + t];
  __syncthreads();
  float acc = b1[t];
#pragma unroll 8
  for (int k = 0; k < 64; ++k) acc += fin[k] * W1[k * 64 + t];
  hid[t] = fmaxf(acc, 0.0f);
  __syncthreads();
  int pos = (int)(fin[0] * 50.0f);
  pos = min(max(pos, 0), 54);
#pragma unroll
  for (int j0 = 0; j0 < 4; ++j0) {
    int j = t + j0 * 64;
    float a = b2[j] + pe[pos * 256 + j];
#pragma unroll 8
    for (int k = 0; k < 64; ++k) a += hid[k] * W2[k * 256 + j];
    x[(size_t)node * 256 + j]  = a;
    xb[(size_t)node * 256 + j] = f2bf(a);
  }
}

// ---------------------------------------------------------------------------
// Degree / normalization
// ---------------------------------------------------------------------------
__global__ void deg_kernel(const int* __restrict__ dst, float* __restrict__ deg, int E) {
  int e = blockIdx.x * blockDim.x + threadIdx.x;
  if (e < E) atomicAdd(&deg[dst[e]], 1.0f);
}

__global__ void dinv_kernel(const float* __restrict__ deg, float* __restrict__ dinv,
                            float* __restrict__ snorm, int n) {
  int i = blockIdx.x * blockDim.x + threadIdx.x;
  if (i >= n) return;
  float di = rsqrtf(deg[i] + 1.0f);
  dinv[i] = di;
  snorm[i] = di * di;
}

__global__ void enorm_kernel(const int* __restrict__ src, const int* __restrict__ dst,
                             const float* __restrict__ dinv, float* __restrict__ enorm, int E) {
  int e = blockIdx.x * blockDim.x + threadIdx.x;
  if (e < E) enorm[e] = dinv[src[e]] * dinv[dst[e]];
}

// ---------------------------------------------------------------------------
// Node GEMM: C[M,Nn] = Abf[M,256] @ Bpacked (+bias)(+relu). 32-row tiles,
// 8 waves/block; each B fragment feeds two WMMAs (two 16-row M-subtiles).
// ---------------------------------------------------------------------------
__global__ __launch_bounds__(256) void gemm_nodes_kernel(
    const __bf16* __restrict__ Ab, const __bf16* __restrict__ Bp,
    const float* __restrict__ bias, float* __restrict__ Cout,
    __bf16* __restrict__ CoutBf, int M, int Nn, int relu) {
  __shared__ __bf16 Ash[32 * 256];
  const int tid  = threadIdx.x;
  const int wave = tid >> 5;
  const int lane = tid & 31;
  const int m0   = blockIdx.x * 32;

  // Stage 32 rows of A into LDS (each thread: 64B = 32 bf16)
  {
    int row = tid >> 3, ch = tid & 7;
    int m = m0 + row; if (m >= M) m = M - 1;
    const uint4* s = (const uint4*)(Ab + (size_t)m * 256 + ch * 32);
    uint4* d = (uint4*)(Ash + row * 256 + ch * 32);
    d[0] = s[0]; d[1] = s[1]; d[2] = s[2]; d[3] = s[3];
  }
  const v16bf* Bpv = (const v16bf*)Bp;
  __builtin_prefetch((const void*)(Bpv + (size_t)(wave * 8) * 32 + lane), 0, 0);
  __syncthreads();

  const int ntiles = Nn >> 4;
  for (int nt = wave; nt < ntiles; nt += 8) {
    v8f acc0 = {}, acc1 = {};
#pragma unroll
    for (int kt = 0; kt < 8; ++kt) {   // K = 256 = 8 * 32
      v16bf b  = Bpv[(size_t)(nt * 8 + kt) * 32 + lane];
      v16bf a0 = load_a_frag(Ash, 256, kt, lane);
      v16bf a1 = load_a_frag(Ash + 16 * 256, 256, kt, lane);
      acc0 = __builtin_amdgcn_wmma_f32_16x16x32_bf16(false, a0, false, b,
                                                     (short)0, acc0, false, false);
      acc1 = __builtin_amdgcn_wmma_f32_16x16x32_bf16(false, a1, false, b,
                                                     (short)0, acc1, false, false);
    }
    int n = nt * 16 + (lane & 15);
    float bv = bias ? bias[n] : 0.0f;
    int mbase = (lane & 16) ? 8 : 0;
#pragma unroll
    for (int r = 0; r < 8; ++r) {
      int m = m0 + mbase + r;
      if (m < M) {
        float v = acc0[r] + bv;
        if (relu) v = fmaxf(v, 0.0f);
        Cout[(size_t)m * Nn + n] = v;
        if (CoutBf) CoutBf[(size_t)m * Nn + n] = f2bf(v);
      }
      int m2 = m + 16;
      if (m2 < M) {
        float v = acc1[r] + bv;
        if (relu) v = fmaxf(v, 0.0f);
        Cout[(size_t)m2 * Nn + n] = v;
        if (CoutBf) CoutBf[(size_t)m2 * Nn + n] = f2bf(v);
      }
    }
  }
}

// ---------------------------------------------------------------------------
// Scatter: agg[dst] += H[src] * enorm  (f32 atomics; H rows are L2 resident)
// ---------------------------------------------------------------------------
__global__ void scatter_kernel(const float* __restrict__ H, const int* __restrict__ src,
                               const int* __restrict__ dst, const float* __restrict__ enorm,
                               float* __restrict__ agg, int E) {
  long long id = (long long)blockIdx.x * blockDim.x + threadIdx.x;
  if (id >= (long long)E * 64) return;
  int e = (int)(id >> 6);
  int c = ((int)id & 63) * 4;
  int s = src[e], d = dst[e];
  float w = enorm[e];
  float4 h = *(const float4*)(H + (size_t)s * 256 + c);
  float* a = agg + (size_t)d * 256 + c;
  atomicAdd(a + 0, h.x * w);
  atomicAdd(a + 1, h.y * w);
  atomicAdd(a + 2, h.z * w);
  atomicAdd(a + 3, h.w * w);
}

// ---------------------------------------------------------------------------
// Combine: x = agg + snorm*H + b  (+relu); writes f32 x and bf16 xb mirrors.
// ---------------------------------------------------------------------------
__global__ void combine_kernel(const float* __restrict__ agg, const float* __restrict__ H,
                               const float* __restrict__ snorm, const float* __restrict__ b,
                               float* __restrict__ xout, __bf16* __restrict__ xbout,
                               int relu, int n_nodes) {
  int i = blockIdx.x * blockDim.x + threadIdx.x;
  if (i >= n_nodes * 256) return;
  int node = i >> 8, f = i & 255;
  float v = agg[i] + snorm[node] * H[i] + b[f];
  if (relu) v = fmaxf(v, 0.0f);
  xout[i]  = v;
  xbout[i] = f2bf(v);
}

// ---------------------------------------------------------------------------
// Head tails (tiny)
// ---------------------------------------------------------------------------
__global__ void ab_kernel(const float* __restrict__ A, const float* __restrict__ B,
                          float* __restrict__ Mab) {
  int i = blockIdx.x * blockDim.x + threadIdx.x;
  if (i >= 512) return;
  int r = i >> 2, c = i & 3;
  float s = 0.0f;
#pragma unroll
  for (int k = 0; k < 4; ++k) s += A[r * 4 + k] * B[k * 4 + c];
  Mab[i] = s;
}

__global__ void predict_kernel(const float* __restrict__ hC, const float* __restrict__ Wc2,
                               const float* __restrict__ bc2, float* __restrict__ outp, int n) {
  int i = blockIdx.x * blockDim.x + threadIdx.x;
  if (i >= n * 6) return;
  int node = i / 6, c = i % 6;
  float s = bc2[c];
#pragma unroll 8
  for (int k = 0; k < 128; ++k) s += hC[(size_t)node * 128 + k] * Wc2[k * 6 + c];
  outp[i] = s;
}

__global__ void box_kernel(const float* __restrict__ hB, const float* __restrict__ Mab,
                           const float* __restrict__ inputs, float* __restrict__ outb, int n) {
  int i = blockIdx.x * blockDim.x + threadIdx.x;
  if (i >= n * 4) return;
  int node = i >> 2, c = i & 3;
  float s = 0.0f;
#pragma unroll 8
  for (int k = 0; k < 128; ++k) s += hB[(size_t)node * 128 + k] * Mab[k * 4 + c];
  outb[i] = tanhf(s) + inputs[(size_t)node * 64 + 1 + c];
}

// ---------------------------------------------------------------------------
// Edge head: 32 edges per block, fully fused.
//   [x[src]|x[dst]] (32x512 bf16 in LDS) @We1(512x256)+relu @We2(256x128)+relu
//   dot We3(128) -> sigmoid.  Each B fragment load feeds 2 WMMAs.
// ---------------------------------------------------------------------------
__global__ __launch_bounds__(256) void edge_head_kernel(
    const __bf16* __restrict__ xb, const int* __restrict__ src, const int* __restrict__ dst,
    const __bf16* __restrict__ We1p, const float* __restrict__ be1,
    const __bf16* __restrict__ We2p, const float* __restrict__ be2,
    const float* __restrict__ We3, const float* __restrict__ be3,
    float* __restrict__ edge_out, int E) {
  __shared__ __bf16 Ash[32 * 512];
  __shared__ __bf16 H1[32 * 256];
  __shared__ float  H2[32 * 128];
  const int tid  = threadIdx.x;
  const int wave = tid >> 5;
  const int lane = tid & 31;
  const int e0   = blockIdx.x * 32;
  const v16bf* B1 = (const v16bf*)We1p;
  const v16bf* B2 = (const v16bf*)We2p;

  // Gather: 8 threads per edge, each copies 64 bf16 (128B) of the 512-wide row.
  {
    int er = tid >> 3;   // 0..31
    int ch = tid & 7;    // 0..7 chunks of 64 bf16
    int e = e0 + er; if (e >= E) e = E - 1;
    int s = src[e], d = dst[e];
    const __bf16* rowp = (ch < 4) ? (xb + (size_t)s * 256 + ch * 64)
                                  : (xb + (size_t)d * 256 + (ch - 4) * 64);
    const uint4* sp = (const uint4*)rowp;
    uint4* dp = (uint4*)(Ash + er * 512 + ch * 64);
#pragma unroll
    for (int i = 0; i < 8; ++i) dp[i] = sp[i];
  }
  // Warm this wave's first B tiles while staging completes.
  __builtin_prefetch((const void*)(B1 + (size_t)(wave * 16) * 32 + lane), 0, 0);
  __syncthreads();

  // GEMM1: 32x512 @ 512x256, 16 n-tiles over 8 waves (2 each), K-loop = 16.
  for (int nt = wave; nt < 16; nt += 8) {
    v8f acc0 = {}, acc1 = {};
#pragma unroll
    for (int kt = 0; kt < 16; ++kt) {
      v16bf b  = B1[(size_t)(nt * 16 + kt) * 32 + lane];
      v16bf a0 = load_a_frag(Ash, 512, kt, lane);
      v16bf a1 = load_a_frag(Ash + 16 * 512, 512, kt, lane);
      acc0 = __builtin_amdgcn_wmma_f32_16x16x32_bf16(false, a0, false, b,
                                                     (short)0, acc0, false, false);
      acc1 = __builtin_amdgcn_wmma_f32_16x16x32_bf16(false, a1, false, b,
                                                     (short)0, acc1, false, false);
    }
    int n = nt * 16 + (lane & 15);
    float bv = be1[n];
    int mbase = (lane & 16) ? 8 : 0;
#pragma unroll
    for (int r = 0; r < 8; ++r) {
      H1[(mbase + r) * 256 + n]      = f2bf(fmaxf(acc0[r] + bv, 0.0f));
      H1[(16 + mbase + r) * 256 + n] = f2bf(fmaxf(acc1[r] + bv, 0.0f));
    }
  }
  __syncthreads();

  // GEMM2: 32x256 @ 256x128, 8 n-tiles, one per wave, K-loop = 8.
  {
    int nt = wave;
    v8f acc0 = {}, acc1 = {};
#pragma unroll
    for (int kt = 0; kt < 8; ++kt) {
      v16bf b  = B2[(size_t)(nt * 8 + kt) * 32 + lane];
      v16bf a0 = load_a_frag(H1, 256, kt, lane);
      v16bf a1 = load_a_frag(H1 + 16 * 256, 256, kt, lane);
      acc0 = __builtin_amdgcn_wmma_f32_16x16x32_bf16(false, a0, false, b,
                                                     (short)0, acc0, false, false);
      acc1 = __builtin_amdgcn_wmma_f32_16x16x32_bf16(false, a1, false, b,
                                                     (short)0, acc1, false, false);
    }
    int n = nt * 16 + (lane & 15);
    float bv = be2[n];
    int mbase = (lane & 16) ? 8 : 0;
#pragma unroll
    for (int r = 0; r < 8; ++r) {
      H2[(mbase + r) * 128 + n]      = fmaxf(acc0[r] + bv, 0.0f);
      H2[(16 + mbase + r) * 128 + n] = fmaxf(acc1[r] + bv, 0.0f);
    }
  }
  __syncthreads();

  // Final 128-dot + sigmoid (one lane per edge; one full wave)
  if (tid < 32) {
    float s = be3[0];
#pragma unroll 8
    for (int j = 0; j < 128; ++j) s += H2[tid * 128 + j] * We3[j];
    int e = e0 + tid;
    if (e < E) edge_out[e] = 1.0f / (1.0f + __expf(-s));
  }
}

// ---------------------------------------------------------------------------
// Host launcher
// ---------------------------------------------------------------------------
extern "C" void kernel_launch(void* const* d_in, const int* in_sizes, int n_in,
                              void* d_out, int out_size, void* d_ws, size_t ws_size,
                              hipStream_t stream) {
  const int N = in_sizes[0] / 64;   // 10000
  const int E = in_sizes[1] / 2;    // 320000

  const float* inputs = (const float*)d_in[0];
  const int*   eidx   = (const int*)d_in[1];
  const int*   src    = eidx;
  const int*   dst    = eidx + E;
  const float* pe  = (const float*)d_in[2];
  const float* W1  = (const float*)d_in[3];  const float* b1  = (const float*)d_in[4];
  const float* W2  = (const float*)d_in[5];  const float* b2  = (const float*)d_in[6];
  const float* Wg1 = (const float*)d_in[7];  const float* bg1 = (const float*)d_in[8];
  const float* Wg2 = (const float*)d_in[9];  const float* bg2 = (const float*)d_in[10];
  const float* Wg3 = (const float*)d_in[11]; const float* bg3 = (const float*)d_in[12];
  const float* Wc1 = (const float*)d_in[13]; const float* bc1 = (const float*)d_in[14];
  const float* Wc2 = (const float*)d_in[15]; const float* bc2 = (const float*)d_in[16];
  const float* Wb1 = (const float*)d_in[17]; const float* bb1 = (const float*)d_in[18];
  const float* Amat = (const float*)d_in[19]; const float* Bmat = (const float*)d_in[20];
  const float* We1 = (const float*)d_in[21]; const float* be1 = (const float*)d_in[22];
  const float* We2 = (const float*)d_in[23]; const float* be2 = (const float*)d_in[24];
  const float* We3 = (const float*)d_in[25]; const float* be3 = (const float*)d_in[26];

  float* out = (float*)d_out;
  float* out_predict = out;                      // [N,6]
  float* out_box     = out + (size_t)N * 6;      // [N,4]
  float* out_edge    = out + (size_t)N * 10;     // [E,1]
  float* out_x       = out + (size_t)N * 10 + E; // [N,256]

  // Workspace bump allocator
  char* w = (char*)d_ws;
  auto alloc = [&](size_t bytes) -> void* {
    void* p = (void*)w;
    w += (bytes + 255) & ~(size_t)255;
    return p;
  };
  __bf16* xb    = (__bf16*)alloc((size_t)N * 256 * 2);
  float*  x     = (float*)alloc((size_t)N * 256 * 4);
  float*  H     = (float*)alloc((size_t)N * 256 * 4);
  float*  agg   = (float*)alloc((size_t)N * 256 * 4);
  float*  deg   = (float*)alloc((size_t)N * 4);
  float*  dinv  = (float*)alloc((size_t)N * 4);
  float*  snorm = (float*)alloc((size_t)N * 4);
  float*  enorm = (float*)alloc((size_t)E * 4);
  __bf16* We1p  = (__bf16*)alloc((size_t)512 * 256 * 2);
  __bf16* We2p  = (__bf16*)alloc((size_t)256 * 128 * 2);
  __bf16* Wg1p  = (__bf16*)alloc((size_t)256 * 256 * 2);
  __bf16* Wg2p  = (__bf16*)alloc((size_t)256 * 256 * 2);
  __bf16* Wg3p  = (__bf16*)alloc((size_t)256 * 256 * 2);
  __bf16* Wc1p  = (__bf16*)alloc((size_t)256 * 128 * 2);
  __bf16* Wb1p  = (__bf16*)alloc((size_t)256 * 128 * 2);
  float*  hC    = (float*)alloc((size_t)N * 128 * 4);
  float*  hB    = (float*)alloc((size_t)N * 128 * 4);
  float*  Mab   = (float*)alloc((size_t)128 * 4 * 4);
  (void)ws_size; (void)n_in; (void)out_size;

  // --- Pack weights (bf16 fragment layout) ---
  pack_b_kernel<<<(16 * 16 * 32 + 255) / 256, 256, 0, stream>>>(We1, We1p, 512, 256);
  pack_b_kernel<<<(8 * 8 * 32 + 255) / 256, 256, 0, stream>>>(We2, We2p, 256, 128);
  pack_b_kernel<<<(16 * 8 * 32 + 255) / 256, 256, 0, stream>>>(Wg1, Wg1p, 256, 256);
  pack_b_kernel<<<(16 * 8 * 32 + 255) / 256, 256, 0, stream>>>(Wg2, Wg2p, 256, 256);
  pack_b_kernel<<<(16 * 8 * 32 + 255) / 256, 256, 0, stream>>>(Wg3, Wg3p, 256, 256);
  pack_b_kernel<<<(8 * 8 * 32 + 255) / 256, 256, 0, stream>>>(Wc1, Wc1p, 256, 128);
  pack_b_kernel<<<(8 * 8 * 32 + 255) / 256, 256, 0, stream>>>(Wb1, Wb1p, 256, 128);
  ab_kernel<<<2, 256, 0, stream>>>(Amat, Bmat, Mab);

  // --- Prep: feat + positional embed -> x, xb ---
  prep_kernel<<<N, 64, 0, stream>>>(inputs, pe, W1, b1, W2, b2, x, xb, N);

  // --- Graph normalization ---
  hipMemsetAsync(deg, 0, (size_t)N * 4, stream);
  deg_kernel<<<(E + 255) / 256, 256, 0, stream>>>(dst, deg, E);
  dinv_kernel<<<(N + 255) / 256, 256, 0, stream>>>(deg, dinv, snorm, N);
  enorm_kernel<<<(E + 255) / 256, 256, 0, stream>>>(src, dst, dinv, enorm, E);

  // --- 3x GCNConv ---
  const int mblocks = (N + 31) / 32;
  const long long sthreads = (long long)E * 64;
  const int sblocks = (int)((sthreads + 255) / 256);
  const __bf16* Wgp[3] = {Wg1p, Wg2p, Wg3p};
  const float*  bgp[3] = {bg1, bg2, bg3};
  for (int l = 0; l < 3; ++l) {
    gemm_nodes_kernel<<<mblocks, 256, 0, stream>>>(xb, Wgp[l], nullptr, H, nullptr,
                                                   N, 256, 0);
    hipMemsetAsync(agg, 0, (size_t)N * 256 * 4, stream);
    scatter_kernel<<<sblocks, 256, 0, stream>>>(H, src, dst, enorm, agg, E);
    float* xtarget = (l == 2) ? out_x : x;
    combine_kernel<<<(N * 256 + 255) / 256, 256, 0, stream>>>(
        agg, H, snorm, bgp[l], xtarget, xb, (l < 2) ? 1 : 0, N);
  }

  // --- cls / box heads ---
  gemm_nodes_kernel<<<mblocks, 256, 0, stream>>>(xb, Wc1p, bc1, hC, nullptr, N, 128, 1);
  gemm_nodes_kernel<<<mblocks, 256, 0, stream>>>(xb, Wb1p, bb1, hB, nullptr, N, 128, 1);
  predict_kernel<<<(N * 6 + 255) / 256, 256, 0, stream>>>(hC, Wc2, bc2, out_predict, N);
  box_kernel<<<(N * 4 + 255) / 256, 256, 0, stream>>>(hB, Mab, inputs, out_box, N);

  // --- edge head (dominant compute) ---
  edge_head_kernel<<<(E + 31) / 32, 256, 0, stream>>>(
      xb, src, dst, We1p, be1, We2p, be2, We3, be3, out_edge, E);
}
